// PositionCloser_2731599200468
// MI455X (gfx1250) — matrix operand order repaired and checked by reading, not compiled, not run on hardware
//
#include <hip/hip_runtime.h>
#include <hip/hip_bf16.h>

typedef __attribute__((ext_vector_type(16))) _Float16 v16h;
typedef __attribute__((ext_vector_type(8)))  float    v8f;

#define HID    64
#define GATES  4
#define CHUNKC 128
#define BB2    16

__device__ __forceinline__ float fast_sigmoid(float x) {
    return __builtin_amdgcn_rcpf(1.0f + __expf(-x));
}
__device__ __forceinline__ float fast_tanh(float x) {
    x = fminf(fmaxf(x, -10.0f), 10.0f);
    float e = __expf(2.0f * x);
    return (e - 1.0f) * __builtin_amdgcn_rcpf(e + 1.0f);
}
__device__ __forceinline__ v16h splat16(_Float16 v) {
    v16h r;
#pragma unroll
    for (int i = 0; i < 16; ++i) r[i] = v;
    return r;
}
__device__ __forceinline__ v8f zero8() {
    v8f r;
#pragma unroll
    for (int i = 0; i < 8; ++i) r[i] = 0.0f;
    return r;
}

// One workgroup = 16 batch rows (one b1 slice), 4 waves; wave w owns hid cols [16w,16w+16).
// gates^T = [W_hh | W_ih | bias] x [h ; x0 ; x1 ; 1] via 3 chained V_WMMA_F32_16X16X32_F16 per gate.
// B operand lives in one LDS tile of 96 K-rows x 16 N-cols per buffer:
//   rows 0..63 = h^T (rewritten each step), row 64 = x0, row 65 = x1,
//   row 66 = ones, rows 67..95 = zeros (written once) -> branch-free fragment loads.
__global__ __launch_bounds__(128, 1)
void PositionCloser_kernel(const int*   __restrict__ inds,
                           const float* __restrict__ p,
                           const float* __restrict__ ls_probs,
                           const float* __restrict__ open_probs,
                           const int*   __restrict__ open_slices,
                           const float* __restrict__ open_hx,
                           const float* __restrict__ W_ih,
                           const float* __restrict__ W_hh,
                           const float* __restrict__ b_ih,
                           const float* __restrict__ b_hh,
                           const float* __restrict__ W_out,
                           const float* __restrict__ b_out,
                           const int*   __restrict__ n_chunks_p,
                           float*       __restrict__ d_out)
{
    __shared__ __align__(32) _Float16 hT[2][96][16];   // [buf][k][row]
    __shared__ float lpbuf[2][16];
    __shared__ float part[2][4][16];

    const int tid  = threadIdx.x;
    const int lane = tid & 31;
    const int wave = tid >> 5;          // 0..3
    const int nrow = lane & 15;         // batch row within WG (== b2)
    const int hi   = lane >> 4;
    const int b1   = blockIdx.x;
    const int row  = b1 * BB2 + nrow;   // global flattened batch row

    const int n_steps = n_chunks_p[0] * CHUNKC;

    // ---- one-time fill of extension rows (both buffers) ----
    // rows 64,65: x (buffer0 step-0 value is x=0, so zero is correct);
    // row 66: ones; rows 67..95: zeros.
    for (int i = tid; i < 2 * 32 * 16; i += 128) {
        int bufi = i >> 9;
        int rr   = (i >> 4) & 31;
        int cc   = i & 15;
        hT[bufi][64 + rr][cc] = (rr == 2) ? (_Float16)1.0f : (_Float16)0.0f;
    }

    // ---- constant A fragments (loaded once, live in VGPRs) ----
    v16h Ah[GATES][2];
    v16h Aext[GATES];
    const int within_gate_col = 16 * wave + nrow;
#pragma unroll
    for (int G = 0; G < GATES; ++G) {
        const int gc = G * HID + within_gate_col;   // gate-matrix row 0..255
#pragma unroll
        for (int kh = 0; kh < 2; ++kh) {
            v16h a;
#pragma unroll
            for (int e = 0; e < 16; ++e) {
                // 16-bit A layout: K(e) = e + (e>=8 ? 8 : 0) + 8*hi (per K-half)
                int k = kh * 32 + e + ((e >= 8) ? 8 : 0) + hi * 8;
                a[e] = (_Float16)W_hh[gc * HID + k];
            }
            Ah[G][kh] = a;
        }
        v16h ax = splat16((_Float16)0.0f);
        if (hi == 0) {                              // ext rows K'=0,1,2 only
            ax[0] = (_Float16)W_ih[gc * 2 + 0];
            ax[1] = (_Float16)W_ih[gc * 2 + 1];
            ax[2] = (_Float16)(b_ih[gc] + b_hh[gc]);
        }
        Aext[G] = ax;
    }

    // ---- per-lane state: c cell + W_out slice, and seed h0 into LDS ----
    float cst[8], wout[8];
#pragma unroll
    for (int r = 0; r < 8; ++r) {
        int j = 16 * wave + r + hi * 8;             // hid column for vgpr r
        cst[r]  = open_hx[(row * 2 + 1) * HID + j];
        wout[r] = W_out[j];
        hT[0][j][nrow] = (_Float16)open_hx[(row * 2 + 0) * HID + j];
    }

    // ---- scalar-chain state (wave 0 lanes 0..15) ----
    float Krow = 0.0f, Crow = 0.0f, r_nd = 1.0f, prev_s = 0.0f;
    float loss = 0.0f, bout = 0.0f;
    if (wave == 0 && hi == 0) {
        float ls = ls_probs[row];
        Krow = open_probs[row] * (2.0f * ls - 1.0f);
        int os = open_slices[row];
        Crow = __logf(p[2 * os]) + __logf(p[2 * os + 1]);
        bout = b_out[0];
    }
    // ---- staging state (wave 3 lanes 0..15) ----
    int posbase = 0;
    float ps0 = 0.0f, ps1 = 0.0f;
    if (wave == 3 && hi == 0) {
        posbase = inds[b1] + nrow;
        float p0 = p[2 * posbase], p1 = p[2 * posbase + 1];
        ps0 = p0; ps1 = p1;                         // chunk-start reference
        lpbuf[0][nrow] = __logf(p0) + __logf(p1);   // x rows already 0 for t=0
    }
    __syncthreads();

    // ================= sequential scan: ONE barrier per step =================
    for (int t = 0; t < n_steps; ++t) {
        const int buf = t & 1;
        const int nb  = buf ^ 1;
        const int kidx = hi * 8;                    // B-fragment K base (table order)

        // branch-free B fragments: h^T, h^T(+32), extension rows (+64)
        v16h Bh0, Bh1, Bx;
#pragma unroll
        for (int e = 0; e < 16; ++e) {
            int k = kidx + e + ((e >= 8) ? 8 : 0);
            Bh0[e] = hT[buf][k][nrow];
            Bh1[e] = hT[buf][k + 32][nrow];
            Bx[e]  = hT[buf][k + 64][nrow];
        }

        v8f acc[GATES];
#pragma unroll
        for (int G = 0; G < GATES; ++G) {
            v8f c = zero8();
            c = __builtin_amdgcn_wmma_f32_16x16x32_f16(false, Ah[G][0], false, Bh0, (short)0, c, false, false);
            c = __builtin_amdgcn_wmma_f32_16x16x32_f16(false, Ah[G][1], false, Bh1, (short)0, c, false, false);
            c = __builtin_amdgcn_wmma_f32_16x16x32_f16(false, Aext[G], false, Bx,  (short)0, c, false, false);
            acc[G] = c;
        }

        // lane-local LSTM cell update (gates i,f,g,o aligned across tiles)
        float hv[8];
        float pd = 0.0f;
#pragma unroll
        for (int r = 0; r < 8; ++r) {
            float ig = fast_sigmoid(acc[0][r]);
            float fg = fast_sigmoid(acc[1][r]);
            float gg = fast_tanh(acc[2][r]);
            float og = fast_sigmoid(acc[3][r]);
            float cv = __builtin_fmaf(fg, cst[r], ig * gg);
            cst[r] = cv;
            float h = og * fast_tanh(cv);
            hv[r] = h;
            pd = __builtin_fmaf(h, wout[r], pd);    // partial h . W_out
        }

        // write h^T into the other buffer (read only after the barrier)
#pragma unroll
        for (int r = 0; r < 8; ++r) {
            hT[nb][16 * wave + r + hi * 8][nrow] = (_Float16)hv[r];
        }
        // fold lane halves of the dot product, publish per-wave partial (double-buffered)
        pd += __shfl_xor(pd, 16, 32);
        if (hi == 0) part[buf][wave][nrow] = pd;

        // wave 0: hoist lp read into phase 1 (staged last iteration)
        float lp = 0.0f;
        if (wave == 0 && hi == 0) lp = lpbuf[buf][nrow];

        // wave 3: stage x and sum-log-p for step t+1 into the next buffer
        if (wave == 3 && hi == 0) {
            int pos = posbase + t + 1;
            float p0 = p[2 * pos], p1 = p[2 * pos + 1];
            if (((t + 1) & (CHUNKC - 1)) == 0) { ps0 = p0; ps1 = p1; }
            hT[nb][64][nrow] = (_Float16)(p0 - ps0);
            hT[nb][65][nrow] = (_Float16)(p1 - ps1);
            lpbuf[nb][nrow]  = __logf(p0) + __logf(p1);
        }

        __syncthreads();

        // wave 0: per-row not-done chain + loss (reads part[buf], safe until barrier t+1)
        if (wave == 0 && hi == 0) {
            float dot = part[buf][0][nrow] + part[buf][1][nrow]
                      + part[buf][2][nrow] + part[buf][3][nrow] + bout;
            float s = fast_sigmoid(dot);
            int c_in = t & (CHUNKC - 1);
            if (c_in != 0) r_nd *= (1.0f - prev_s);
            float prob = (c_in == 0) ? s : s * r_nd;
            prev_s = s;
            loss = __builtin_fmaf(Krow * prob, lp - Crow, loss);
        }
    }

    // final loss reduction: 16 row-lanes of wave 0 -> one atomicAdd per WG
    if (wave == 0 && hi == 0) {
        loss += __shfl_xor(loss, 8, 32);
        loss += __shfl_xor(loss, 4, 32);
        loss += __shfl_xor(loss, 2, 32);
        loss += __shfl_xor(loss, 1, 32);
        if (nrow == 0) atomicAdd(d_out, loss);
    }
}

extern "C" void kernel_launch(void* const* d_in, const int* in_sizes, int n_in,
                              void* d_out, int out_size, void* d_ws, size_t ws_size,
                              hipStream_t stream) {
    (void)n_in; (void)out_size; (void)d_ws; (void)ws_size;
    const int B1 = in_sizes[0];                 // 64
    hipMemsetAsync(d_out, 0, sizeof(float), stream);
    dim3 grid(B1), block(128);
    PositionCloser_kernel<<<grid, block, 0, stream>>>(
        (const int*)  d_in[0],   // inds
        (const float*)d_in[1],   // p
        (const float*)d_in[2],   // ls_probs
        (const float*)d_in[3],   // open_probs
        (const int*)  d_in[4],   // open_slices
        (const float*)d_in[5],   // open_hx
        (const float*)d_in[6],   // W_ih
        (const float*)d_in[7],   // W_hh
        (const float*)d_in[8],   // b_ih
        (const float*)d_in[9],   // b_hh
        (const float*)d_in[10],  // W_out
        (const float*)d_in[11],  // b_out
        (const int*)  d_in[12],  // n_chunks
        (float*)d_out);
}
